// ProbAttention_55087250538864
// MI455X (gfx1250) — matrix operand order repaired and checked by reading, not compiled, not run on hardware
//
#include <hip/hip_runtime.h>
#include <hip/hip_bf16.h>

// ---------------- geometry ----------------
#define BATCH   16
#define CCH     512        // 2*Cin, GEMM K for projections
#define LTOK    4096       // 64*64 spatial tokens
#define DM      512        // model dim
#define USEL    450        // top-k / sample count
#define UPAD    464        // 29 tiles of 16
#define UTILES  29
#define SCALE   0.04419417382415922f   // 1/sqrt(512)

typedef __attribute__((ext_vector_type(16))) __bf16 v16bf;
typedef __attribute__((ext_vector_type(8)))  float  v8f;

union FragAB { v16bf v; unsigned u[8]; __bf16 h[16]; };
union FragC  { v8f   v; float    f[8]; };

// ISA 16-bit A-matrix 16x32 layout: lane holds row m=lane&15, half h=lane>>4.
// VGPR v holds K-pair at ((v>>2)<<4) | (h<<3) | ((v&3)<<1).
__device__ __forceinline__ int akoff(int v, int h) {
    return ((v >> 2) << 4) | (h << 3) | ((v & 3) << 1);
}
// ISA 16-bit B-matrix 32x16 layout: lane holds col n=lane&15, half h=lane>>4.
// VGPR v holds K-pair at ((v>>2)<<4) | (h<<3) | ((v&3)<<1) (same as A).
__device__ __forceinline__ int bkoff(int v, int h) {
    return ((v >> 2) << 4) | (h << 3) | ((v & 3) << 1);
}

// rowPtr already points at this lane's row (src row-major, K contiguous).
__device__ __forceinline__ void loadA(FragAB& f, const __bf16* rowPtr, int h) {
    const unsigned short* p = (const unsigned short*)rowPtr;
#pragma unroll
    for (int v = 0; v < 8; ++v) f.u[v] = *(const unsigned*)(p + akoff(v, h));
}
__device__ __forceinline__ void loadB(FragAB& f, const __bf16* rowPtr, int h) {
    const unsigned short* p = (const unsigned short*)rowPtr;
#pragma unroll
    for (int v = 0; v < 8; ++v) f.u[v] = *(const unsigned*)(p + bkoff(v, h));
}
__device__ __forceinline__ void wmma_bf16(FragC& c, const FragAB& a, const FragAB& b) {
    c.v = __builtin_amdgcn_wmma_f32_16x16x32_bf16(false, a.v, false, b.v,
                                                  (short)0, c.v, false, false);
}

// ---------------- K1: weights f32 -> bf16 ----------------
__global__ void pack_w(const float* Wq, const float* Wk, const float* Wv, __bf16* Wb) {
    int i = blockIdx.x * 256 + threadIdx.x;          // 3*512*512 total
    const float* src = (i < 262144) ? Wq : (i < 524288) ? Wk : Wv;
    Wb[i] = (__bf16)src[i & 262143];
}

// ---------------- K2: Xt[b][wh][c] = concat(x1,x2)[b][c][wh] as bf16 ----------------
__global__ void pack_xt(const float* in1, const float* in2, __bf16* Xt) {
    __shared__ float tile[16][17];
    int b = blockIdx.z, wh0 = blockIdx.x * 16, c0 = blockIdx.y * 16;
    int tx = threadIdx.x & 15, ty = threadIdx.x >> 4;
    int c = c0 + ty;
    const float* src = (c < 256) ? (in1 + ((size_t)b * 256 + c) * LTOK)
                                 : (in2 + ((size_t)b * 256 + (c - 256)) * LTOK);
    tile[ty][tx] = src[wh0 + tx];
    __syncthreads();
    Xt[((size_t)b * LTOK + wh0 + ty) * CCH + c0 + tx] = (__bf16)tile[tx][ty];
}

// ---------------- K3: projection GEMM  Y[o,wh] = sum_c W[o,c]*X[c,wh] + bias[o] ----------------
// Output buffer (o-major 512x4096) IS the (L,D) view of q/k/v.
__global__ __launch_bounds__(256) void proj_gemm(
        const __bf16* Wb, const float* bq, const float* bk, const float* bv,
        const __bf16* Xt, __bf16* Qb, __bf16* Kb, __bf16* Vb) {
    int z = blockIdx.z, b = z / 3, pr = z % 3;
    const __bf16* W   = Wb + pr * 512 * 512;
    const float* bias = (pr == 0) ? bq : (pr == 1) ? bk : bv;
    __bf16* Out = ((pr == 0) ? Qb : (pr == 1) ? Kb : Vb) + (size_t)b * LTOK * DM;
    const __bf16* X = Xt + (size_t)b * LTOK * CCH;

    int lane = threadIdx.x & 31, w = threadIdx.x >> 5;
    int h = lane >> 4, lo = lane & 15;
    int wm = w & 1, wn = w >> 1;                 // 2(M) x 4(N) waves
    int m0 = blockIdx.y * 64 + wm * 32;          // o
    int n0 = blockIdx.x * 128 + wn * 32;         // wh

    FragC acc[2][2];
#pragma unroll
    for (int i = 0; i < 2; ++i)
#pragma unroll
        for (int j = 0; j < 2; ++j)
#pragma unroll
            for (int t = 0; t < 8; ++t) acc[i][j].f[t] = 0.0f;

    for (int k0 = 0; k0 < CCH; k0 += 32) {
        FragAB a[2], bb[2];
        loadA(a[0], W + (m0 + lo) * CCH + k0, h);
        loadA(a[1], W + (m0 + 16 + lo) * CCH + k0, h);
        loadB(bb[0], X + (n0 + lo) * CCH + k0, h);
        loadB(bb[1], X + (n0 + 16 + lo) * CCH + k0, h);
#pragma unroll
        for (int i = 0; i < 2; ++i)
#pragma unroll
            for (int j = 0; j < 2; ++j) wmma_bf16(acc[i][j], a[i], bb[j]);
    }
#pragma unroll
    for (int i = 0; i < 2; ++i)
#pragma unroll
        for (int j = 0; j < 2; ++j)
#pragma unroll
            for (int v = 0; v < 8; ++v) {
                int o = m0 + i * 16 + v + 8 * h;
                int wh = n0 + j * 16 + lo;
                Out[(size_t)o * LTOK + wh] = (__bf16)(acc[i][j].f[v] + bias[o]);
            }
}

// ---------------- K4: Vt[b][d][l] = Vb[b][l][d] ----------------
__global__ void vtrans(const __bf16* Vb, __bf16* Vt) {
    __shared__ __bf16 tile[16][17];
    int b = blockIdx.z, l0 = blockIdx.x * 16, d0 = blockIdx.y * 16;
    int tx = threadIdx.x & 15, ty = threadIdx.x >> 4;
    tile[ty][tx] = Vb[((size_t)b * LTOK + l0 + ty) * DM + d0 + tx];
    __syncthreads();
    Vt[((size_t)b * DM + d0 + ty) * LTOK + l0 + tx] = tile[tx][ty];
}

// ---------------- K5/K6: deterministic column mean of V ----------------
__global__ void vmean_part(const __bf16* Vb, float* pbuf) {
    int b = blockIdx.x, ch = blockIdx.y, d = threadIdx.x;  // 512 threads
    const __bf16* Vp = Vb + (size_t)b * LTOK * DM;
    float s = 0.0f;
    for (int l = ch * 512; l < ch * 512 + 512; ++l) s += (float)Vp[(size_t)l * DM + d];
    pbuf[((size_t)b * 8 + ch) * DM + d] = s;
}
__global__ void vmean_fin(const float* pbuf, float* vmean) {
    int b = blockIdx.x, d = threadIdx.x;
    float s = 0.0f;
#pragma unroll
    for (int c = 0; c < 8; ++c) s += pbuf[((size_t)b * 8 + c) * DM + d];
    vmean[b * DM + d] = s * (1.0f / 4096.0f);
}

// ---------------- K7: gather K_sample (pad rows zeroed; excluded from M anyway) ----------------
__global__ void ks_gather(const __bf16* Kb, const int* idx, __bf16* Ks) {
    int b = blockIdx.x, st = blockIdx.y;
    for (int t = threadIdx.x; t < 16 * DM; t += 256) {
        int s = st * 16 + (t >> 9), d = t & 511;
        __bf16 val = (__bf16)0.0f;
        if (s < USEL) val = Kb[((size_t)b * LTOK + idx[s]) * DM + d];
        Ks[((size_t)b * UPAD + s) * DM + d] = val;
    }
}

// ---------------- K8: QKs + fused M = rowmax - rowsum/L ----------------
__global__ __launch_bounds__(128) void qks_m(const __bf16* Qb, const __bf16* Ks, float* Mout) {
    int b = blockIdx.y;
    int lane = threadIdx.x & 31, w = threadIdx.x >> 5;
    int h = lane >> 4, lo = lane & 15;
    int l0 = blockIdx.x * 64 + w * 16;
    const __bf16* Q = Qb + (size_t)b * LTOK * DM;
    const __bf16* K = Ks + (size_t)b * UPAD * DM;

    FragAB a[16];                       // cache full 16 x 512 A rows
    const __bf16* rowA = Q + (size_t)(l0 + lo) * DM;
#pragma unroll
    for (int kt = 0; kt < 16; ++kt) loadA(a[kt], rowA + kt * 32, h);

    float rmax[8], rsum[8];
#pragma unroll
    for (int v = 0; v < 8; ++v) { rmax[v] = -3.0e38f; rsum[v] = 0.0f; }

    for (int st = 0; st < UTILES; ++st) {
        int s0 = st * 16;
        const __bf16* rowB = K + (size_t)(s0 + lo) * DM;
        FragC c;
#pragma unroll
        for (int t = 0; t < 8; ++t) c.f[t] = 0.0f;
#pragma unroll
        for (int kt = 0; kt < 16; ++kt) {
            FragAB bb; loadB(bb, rowB + kt * 32, h);
            wmma_bf16(c, a[kt], bb);
        }
        bool valid = (s0 + lo) < USEL;     // mask padded sample columns
#pragma unroll
        for (int v = 0; v < 8; ++v)
            if (valid) { rsum[v] += c.f[v]; rmax[v] = fmaxf(rmax[v], c.f[v]); }
    }
    // reduce across 16 lanes sharing the same row set
#pragma unroll
    for (int v = 0; v < 8; ++v) {
#pragma unroll
        for (int msk = 1; msk <= 8; msk <<= 1) {
            rmax[v] = fmaxf(rmax[v], __shfl_xor(rmax[v], msk, 32));
            rsum[v] += __shfl_xor(rsum[v], msk, 32);
        }
    }
    if (lo == 0)
#pragma unroll
        for (int v = 0; v < 8; ++v)
            Mout[b * LTOK + l0 + v + 8 * h] = rmax[v] - rsum[v] * (1.0f / 4096.0f);
}

// ---------------- K9: deterministic top-450 selection ----------------
__device__ __forceinline__ int block_exscan(int val) {
    static __shared__ int wsum[32];
    int tid = threadIdx.x, lane = tid & 31, wid = tid >> 5;
    __syncthreads();
    int x = val;
#pragma unroll
    for (int d = 1; d < 32; d <<= 1) { int y = __shfl_up(x, d, 32); if (lane >= d) x += y; }
    if (lane == 31) wsum[wid] = x;
    __syncthreads();
    if (wid == 0) {
        int wv = wsum[lane];
#pragma unroll
        for (int d = 1; d < 32; d <<= 1) { int y = __shfl_up(wv, d, 32); if (lane >= d) wv += y; }
        wsum[lane] = wv;
    }
    __syncthreads();
    int base = (wid > 0) ? wsum[wid - 1] : 0;
    return base + x - val;   // exclusive prefix
}

__global__ __launch_bounds__(1024) void topk_sel(const float* Mbuf, int* topk) {
    int b = blockIdx.x, tid = threadIdx.x;
    const float* Mb = Mbuf + b * LTOK;
    unsigned key[4];
#pragma unroll
    for (int j = 0; j < 4; ++j) {
        unsigned bits = __float_as_uint(Mb[tid * 4 + j]);
        key[j] = (bits & 0x80000000u) ? ~bits : (bits | 0x80000000u);  // order-preserving
    }
    __shared__ int cnt;
    unsigned t = 0;
    for (int bit = 31; bit >= 0; --bit) {           // radix threshold search
        unsigned cand = t | (1u << bit);
        if (tid == 0) cnt = 0;
        __syncthreads();
        int lc = 0;
#pragma unroll
        for (int j = 0; j < 4; ++j) lc += (key[j] >= cand) ? 1 : 0;
        if (lc) atomicAdd(&cnt, lc);
        __syncthreads();
        if (cnt >= USEL) t = cand;
        __syncthreads();
    }
    int lg = 0, le = 0;
#pragma unroll
    for (int j = 0; j < 4; ++j) { lg += (key[j] > t); le += (key[j] == t); }
    if (tid == 0) cnt = 0;
    __syncthreads();
    if (lg) atomicAdd(&cnt, lg);
    __syncthreads();
    int G = cnt;
    int offg = block_exscan(lg);
    int offe = block_exscan(le);
    int pg = offg, pe = G + offe;
    int* tb = topk + b * UPAD;
#pragma unroll
    for (int j = 0; j < 4; ++j) {
        int l = tid * 4 + j;
        if (key[j] > t) tb[pg++] = l;
        else if (key[j] == t) { if (pe < USEL) tb[pe] = l; pe++; }
    }
    __syncthreads();
    if (tid < UPAD - USEL) tb[USEL + tid] = tb[0];   // pad to 464
}

// ---------------- K10: gather Q_reduce ----------------
__global__ void qr_gather(const __bf16* Qb, const int* topk, __bf16* Qr) {
    int b = blockIdx.x, st = blockIdx.y;
    for (int t = threadIdx.x; t < 16 * DM; t += 256) {
        int u = st * 16 + (t >> 9), d = t & 511;
        Qr[((size_t)b * UPAD + u) * DM + d] =
            Qb[((size_t)b * LTOK + topk[b * UPAD + u]) * DM + d];
    }
}

// ---------------- K11: scores pass 1 (store bf16 S, online row max / sumexp) ----------------
__global__ __launch_bounds__(128) void scores_pass1(
        const __bf16* Qr, const __bf16* Kb, __bf16* S, float* rowmax, float* rowsum) {
    int b = blockIdx.y;
    int lane = threadIdx.x & 31, w = threadIdx.x >> 5;
    int h = lane >> 4, lo = lane & 15;
    int ut = blockIdx.x * 4 + w;
    if (ut >= UTILES) return;
    int u0 = ut * 16;
    const __bf16* Q = Qr + (size_t)b * UPAD * DM;
    const __bf16* K = Kb + (size_t)b * LTOK * DM;

    FragAB a[16];
    const __bf16* rowA = Q + (size_t)(u0 + lo) * DM;
#pragma unroll
    for (int kt = 0; kt < 16; ++kt) loadA(a[kt], rowA + kt * 32, h);

    float rmax[8], rsum[8];
#pragma unroll
    for (int v = 0; v < 8; ++v) { rmax[v] = -3.0e38f; rsum[v] = 0.0f; }

    for (int lt = 0; lt < 256; ++lt) {
        int l0 = lt * 16;
        const __bf16* rowB = K + (size_t)(l0 + lo) * DM;
        FragC c;
#pragma unroll
        for (int t = 0; t < 8; ++t) c.f[t] = 0.0f;
#pragma unroll
        for (int kt = 0; kt < 16; ++kt) {
            FragAB bb; loadB(bb, rowB + kt * 32, h);
            wmma_bf16(c, a[kt], bb);
        }
#pragma unroll
        for (int v = 0; v < 8; ++v) {
            __bf16 q16 = (__bf16)(c.f[v] * SCALE);
            float vq = (float)q16;          // use the quantized value consistently
            S[(size_t)(b * UPAD + u0 + v + 8 * h) * LTOK + l0 + lo] = q16;
            float nm = fmaxf(rmax[v], vq);
            rsum[v] = rsum[v] * __expf(rmax[v] - nm) + __expf(vq - nm);
            rmax[v] = nm;
        }
    }
#pragma unroll
    for (int v = 0; v < 8; ++v) {
        float rm = rmax[v];
#pragma unroll
        for (int msk = 1; msk <= 8; msk <<= 1) rm = fmaxf(rm, __shfl_xor(rm, msk, 32));
        float rs = rsum[v] * __expf(rmax[v] - rm);
#pragma unroll
        for (int msk = 1; msk <= 8; msk <<= 1) rs += __shfl_xor(rs, msk, 32);
        if (lo == 0) {
            rowmax[b * UPAD + u0 + v + 8 * h] = rm;
            rowsum[b * UPAD + u0 + v + 8 * h] = rs;
        }
    }
}

// ---------------- K12: context = broadcast of vmean ----------------
__global__ void bcast_ctx(const float* vmean, float* out) {
    unsigned i = blockIdx.x * 256u + threadIdx.x;     // 16*4096*512 = 2^25 elements
    out[i] = vmean[((i >> 21) << 9) | (i & 511)];
}

// ---------------- K13: upd = softmax(S) @ V, scattered into context rows ----------------
__global__ __launch_bounds__(256) void upd_scatter(
        const __bf16* S, const __bf16* Vt, const float* rowmax, const float* rowsum,
        const int* topk, float* out) {
    int ut = blockIdx.x, b = blockIdx.y;
    int u0 = ut * 16;
    int lane = threadIdx.x & 31, w = threadIdx.x >> 5;  // 8 waves, d-slice of 64 each
    int h = lane >> 4, lo = lane & 15;
    int d0 = w * 64;
    const __bf16* Sb = S + (size_t)(b * UPAD + u0) * LTOK;
    const __bf16* V  = Vt + (size_t)b * DM * LTOK;
    float rmaxm = rowmax[b * UPAD + u0 + lo];           // row for this lane's A rows

    FragC acc[4];
#pragma unroll
    for (int n = 0; n < 4; ++n)
#pragma unroll
        for (int t = 0; t < 8; ++t) acc[n].f[t] = 0.0f;

    for (int l0 = 0; l0 < LTOK; l0 += 32) {
        FragAB a;
        const unsigned short* srow = (const unsigned short*)(Sb + (size_t)lo * LTOK + l0);
#pragma unroll
        for (int v = 0; v < 8; ++v) {
            union { unsigned u; __bf16 hh[2]; } pu;
            pu.u = *(const unsigned*)(srow + akoff(v, h));
            float p0 = __expf((float)pu.hh[0] - rmaxm);
            float p1 = __expf((float)pu.hh[1] - rmaxm);
            a.h[2 * v]     = (__bf16)p0;
            a.h[2 * v + 1] = (__bf16)p1;
        }
#pragma unroll
        for (int nt = 0; nt < 4; ++nt) {
            FragAB bb;
            loadB(bb, V + (size_t)(d0 + nt * 16 + lo) * LTOK + l0, h);
            wmma_bf16(acc[nt], a, bb);
        }
    }
#pragma unroll
    for (int v = 0; v < 8; ++v) {
        int u = u0 + v + 8 * h;
        if (u < USEL) {
            int row = topk[b * UPAD + u];
            float inv = 1.0f / rowsum[b * UPAD + u];
#pragma unroll
            for (int nt = 0; nt < 4; ++nt)
                out[((size_t)b * LTOK + row) * DM + d0 + nt * 16 + lo] = acc[nt].f[v] * inv;
        }
    }
}

// ---------------- host ----------------
extern "C" void kernel_launch(void* const* d_in, const int* in_sizes, int n_in,
                              void* d_out, int out_size, void* d_ws, size_t ws_size,
                              hipStream_t stream) {
    const float* in1 = (const float*)d_in[0];
    const float* in2 = (const float*)d_in[1];
    const float* Wq  = (const float*)d_in[2];
    const float* bq  = (const float*)d_in[3];
    const float* Wk  = (const float*)d_in[4];
    const float* bk  = (const float*)d_in[5];
    const float* Wv  = (const float*)d_in[6];
    const float* bv  = (const float*)d_in[7];
    const int*   idx = (const int*)d_in[8];
    float* out = (float*)d_out;

    char* p = (char*)d_ws;
    auto alloc = [&](size_t n) { char* r = p; p += (n + 255) & ~(size_t)255; return r; };
    __bf16* Wb   = (__bf16*)alloc((size_t)3 * 512 * 512 * 2);
    __bf16* Xt   = (__bf16*)alloc((size_t)BATCH * LTOK * CCH * 2);   // reused as S
    __bf16* Qb   = (__bf16*)alloc((size_t)BATCH * LTOK * DM * 2);
    __bf16* Kb   = (__bf16*)alloc((size_t)BATCH * LTOK * DM * 2);
    __bf16* Vb   = (__bf16*)alloc((size_t)BATCH * LTOK * DM * 2);
    __bf16* Vt   = (__bf16*)alloc((size_t)BATCH * DM * LTOK * 2);
    __bf16* Ks   = (__bf16*)alloc((size_t)BATCH * UPAD * DM * 2);
    __bf16* Qr   = (__bf16*)alloc((size_t)BATCH * UPAD * DM * 2);
    float* Mbuf  = (float*)alloc((size_t)BATCH * LTOK * 4);
    int*   topk  = (int*)alloc((size_t)BATCH * UPAD * 4);
    float* vmean = (float*)alloc((size_t)BATCH * DM * 4);
    float* pbuf  = (float*)alloc((size_t)BATCH * 8 * DM * 4);
    float* rmaxb = (float*)alloc((size_t)BATCH * UPAD * 4);
    float* rsumb = (float*)alloc((size_t)BATCH * UPAD * 4);
    __bf16* Sbuf = Xt;   // S (16*464*4096 bf16 = 60.8MB) fits in Xt region (67MB)

    pack_w<<<3072, 256, 0, stream>>>(Wq, Wk, Wv, Wb);
    pack_xt<<<dim3(256, 32, BATCH), 256, 0, stream>>>(in1, in2, Xt);
    proj_gemm<<<dim3(32, 8, 48), 256, 0, stream>>>(Wb, bq, bk, bv, Xt, Qb, Kb, Vb);
    vtrans<<<dim3(256, 32, BATCH), 256, 0, stream>>>(Vb, Vt);
    vmean_part<<<dim3(BATCH, 8), 512, 0, stream>>>(Vb, pbuf);
    vmean_fin<<<BATCH, 512, 0, stream>>>(pbuf, vmean);
    ks_gather<<<dim3(BATCH, UTILES), 256, 0, stream>>>(Kb, idx, Ks);
    qks_m<<<dim3(64, BATCH), 128, 0, stream>>>(Qb, Ks, Mbuf);
    topk_sel<<<BATCH, 1024, 0, stream>>>(Mbuf, topk);
    qr_gather<<<dim3(BATCH, UTILES), 256, 0, stream>>>(Qb, topk, Qr);
    scores_pass1<<<dim3(8, BATCH), 128, 0, stream>>>(Qr, Kb, Sbuf, rmaxb, rsumb);
    bcast_ctx<<<131072, 256, 0, stream>>>(vmean, out);
    upd_scatter<<<dim3(UTILES, BATCH), 256, 0, stream>>>(Sbuf, Vt, rmaxb, rsumb, topk, out);

    (void)in_sizes; (void)n_in; (void)out_size; (void)ws_size;
}
